// RQVAE_20014547599811
// MI455X (gfx1250) — matrix operand order repaired and checked by reading, not compiled, not run on hardware
//
#include <hip/hip_runtime.h>
#include <hip/hip_bf16.h>
#include <stddef.h>
#include <stdint.h>

// Problem dims (fixed by the reference)
#define RQ_B 16
#define RQ_W 2048
#define RQ_N (RQ_B * RQ_W)   // 32768 rows
#define RQ_C 512             // feature dim
#define RQ_K 1024            // codes per stage
#define RQ_Q 8               // quantizer stages

#define ROWS_PER_BLOCK 64    // 4 wave-pairs x 16 rows
#define THREADS 256          // 8 waves

// LDS B-buffer rows padded 256 -> 264 elements (528 B = 132 dwords; lane j ->
// banks 4j..4j+3, conflict-free b128 reads). TDM pad_enable makes this pad.
#define BROW 264

typedef __attribute__((ext_vector_type(16))) __bf16 v16bf;
typedef __attribute__((ext_vector_type(8)))  float  v8f;
typedef __attribute__((ext_vector_type(4)))  unsigned int v4u;
typedef __attribute__((ext_vector_type(8)))  int v8i;
typedef __attribute__((ext_vector_type(4)))  int v4i;

// Distance between hi and lo codebook planes, in bf16 elements (contiguous ws)
#define LO_ELEMS ((unsigned)RQ_Q * RQ_K * RQ_C)   // 4194304

__device__ __forceinline__ v8f wmma_bf16(v16bf a, v16bf b, v8f c) {
    return __builtin_amdgcn_wmma_f32_16x16x32_bf16(
        /*neg_a=*/false, a, /*neg_b=*/false, b,
        /*c_mod=*/(short)0, c, /*reuse_a=*/false, /*reuse_b=*/false);
}

// Issue one TDM load: 3D tile (x=256 elems, y=16 rows stride 512,
// z=2 planes stride LO_ELEMS) from global bf16 -> LDS, with LDS padding of
// 4 DWORDs after every 128 DWORDs (one 512-B row) => 264-element LDS rows.
// Tracked by TENSORcnt.  ISA 8.3-8.6 D# bitfields.
__device__ __forceinline__ void tdm_issue(const __bf16* gtile,
                                          unsigned lds_byte) {
    const uint64_t ga = (uint64_t)(size_t)gtile;
    v4u g0;
    g0[0] = 1u;                               // count=1, user mode
    g0[1] = lds_byte;                         // lds_addr
    g0[2] = (unsigned)ga;                     // global_addr[31:0]
    g0[3] = (unsigned)(ga >> 32) | 0x80000000u;  // global_addr[56:32] | type=2
    v8i g1;
    g1[0] = (int)((1u << 16) |                // data_size = 2 bytes
                  (1u << 20) |                // pad_enable
                  (6u << 22) |                // pad_interval: 128 dwords
                  (3u << 25));                // pad_amount: 4 dwords (16 B)
    g1[1] = (int)(256u << 16);                // tensor_dim0 = 256 (bits 63:48)
    g1[2] = (int)(16u << 16);                 // tensor_dim1 = 16  (bits 95:80)
    g1[3] = (int)(256u << 16);                // tile_dim0 = 256 (bits 127:112)
    g1[4] = (int)(16u | (2u << 16));          // tile_dim1 = 16, tile_dim2 = 2
    g1[5] = 512;                              // tensor_dim0_stride[31:0]
    g1[6] = 0;                                // stride0[47:32]|stride1[15:0]=0
    g1[7] = (int)(LO_ELEMS >> 16);            // tensor_dim1_stride[47:16]
    v4i g2 = {2, 0, 0, 0};                    // tensor_dim2 = 2
    v4i g3 = {0, 0, 0, 0};
    v8i g4 = {0, 0, 0, 0, 0, 0, 0, 0};        // unused (6-arg toolchain form)
    __builtin_amdgcn_tensor_load_to_lds(g0, g1, g2, g3, g4, 0);
}

// ---------------------------------------------------------------------------
// Prep: split fp32 codebooks into bf16 hi/lo (error-compensated pair)
// ---------------------------------------------------------------------------
__global__ __launch_bounds__(256) void rq_split_kernel(
    const float* __restrict__ cb, __bf16* __restrict__ h,
    __bf16* __restrict__ l, int n) {
    int i = blockIdx.x * 256 + threadIdx.x;
    if (i < n) {
        float x = cb[i];
        __bf16 hh = (__bf16)x;
        h[i] = hh;
        l[i] = (__bf16)(x - (float)hh);
    }
}

// Prep: per-code squared norms, one wave32 per code (coalesced)
__global__ __launch_bounds__(256) void rq_norm_kernel(
    const float* __restrict__ cb, float* __restrict__ cnorm, int ncodes) {
    int lane = threadIdx.x & 31;
    int code = (blockIdx.x * 256 + threadIdx.x) >> 5;
    if (code >= ncodes) return;
    const float* p = cb + (size_t)code * RQ_C;
    float s = 0.f;
    #pragma unroll
    for (int c = lane; c < RQ_C; c += 32) {
        float v = p[c];
        s += v * v;
    }
    #pragma unroll
    for (int m = 16; m >= 1; m >>= 1) s += __shfl_xor(s, m, 32);
    if (lane == 0) cnorm[code] = s;
}

__global__ void rq_zero_kernel(double* acc) { *acc = 0.0; }

// ---------------------------------------------------------------------------
// Fused per-stage kernel, K-split wave pairs.
// Block = 8 waves = 4 pairs; pair p owns rows [p*16, p*16+16). Within a pair,
// wave parity P holds resident bf16 hi/lo A-fragments for K-half P only
// (128 VGPRs => whole working set fits the 256-VGPR file: no spills).
// The K=1024 codebook is swept in 64 chunks of 16 codes; each chunk (hi+lo,
// full K) is DMA'd global->LDS by two TDM descriptors (wave 0 only) into a
// double buffer shared by all 8 waves. Both parities compute their K-half
// partial of the same chunk each iteration; the even wave's 16x16 partial is
// combined by the odd wave through an LDS exchange tile, which owns argmin.
// ---------------------------------------------------------------------------
__global__ __launch_bounds__(THREADS) void rq_stage_kernel(
    const float*  __restrict__ cb_f32,   // [K, C] fp32 codebook (this stage)
    const __bf16* __restrict__ cbh,      // [K, C] bf16 hi (lo at +LO_ELEMS)
    const float*  __restrict__ cnorm,    // [K] ||c||^2
    float* __restrict__ resid,           // [N, C] residual (read+write)
    double* __restrict__ sumsq) {        // loss accumulator
    // [buf][K-half][hi/lo][code-in-chunk][padded 256-K]
    __shared__ __bf16 Bbuf[2][2][2][16][BROW];       // 66 KB
    __shared__ float  ex[4][16][16];                 // pair partial exchange
    __shared__ int    rowIdx[ROWS_PER_BLOCK];
    __shared__ float  wsum[8];

    const int tid  = threadIdx.x;
    const int lane = tid & 31;
    const int wave = tid >> 5;           // 0..7
    const int pair = wave >> 1;          // 0..3
    const int P    = wave & 1;           // K-half parity
    const int half = lane >> 4;          // 16-lane half
    const int l16  = lane & 15;
    const int row0 = blockIdx.x * ROWS_PER_BLOCK;

    // ---- Phase 1: A fragments for this wave's K-half (8 k-steps) ---------
    // A k order used here (contiguous, same as B): row = lane%16; lanes 0-15 hold K = 0..15,
    // lanes 16-31 hold K = 16..31 (per 32-wide K step).
    const float* arow = resid + (size_t)(row0 + pair * 16 + l16) * RQ_C;
    v16bf a_hi[8], a_lo[8];
    #pragma unroll
    for (int ks = 0; ks < 8; ++ks) {
        const int k0 = (P * 8 + ks) * 32 + half * 16;
        float4 f0 = *(const float4*)(arow + k0);
        float4 f1 = *(const float4*)(arow + k0 + 4);
        float4 f2 = *(const float4*)(arow + k0 + 8);
        float4 f3 = *(const float4*)(arow + k0 + 12);
        float t[16] = {f0.x, f0.y, f0.z, f0.w, f1.x, f1.y, f1.z, f1.w,
                       f2.x, f2.y, f2.z, f2.w, f3.x, f3.y, f3.z, f3.w};
        #pragma unroll
        for (int e = 0; e < 16; ++e) {
            __bf16 hh = (__bf16)t[e];
            a_hi[ks][e] = hh;
            a_lo[ks][e] = (__bf16)(t[e] - (float)hh);
        }
    }

    float vmin[8];
    int   vidx[8];
    #pragma unroll
    for (int v = 0; v < 8; v++) { vmin[v] = 3.4e38f; vidx[v] = 0; }

    // Chunk cc -> buffer cc&1; two TDM descriptors (K-half 0 and 1).
    auto issue_chunk = [&](int cc) {
        const int b = cc & 1;
        const __bf16* g = cbh + (size_t)(cc * 16) * RQ_C;
        tdm_issue(g,       (unsigned)(size_t)&Bbuf[b][0][0][0][0]);
        tdm_issue(g + 256, (unsigned)(size_t)&Bbuf[b][1][0][0][0]);
    };

    if (wave == 0) {
        issue_chunk(0);
        issue_chunk(1);
    }

    for (int cc = 0; cc < 64; ++cc) {
        if (wave == 0) {   // oldest in-flight chunk (2 descriptors) done
            if (cc < 63) {
                __builtin_amdgcn_s_wait_tensorcnt(2);
            } else {
                __builtin_amdgcn_s_wait_tensorcnt(0);
            }
        }
        __syncthreads();   // chunk cc landed + visible to all waves

        const int b = cc & 1;
        // B k order used here (contiguous, same as A): column n = lane%16; lanes 0-15 hold K=0..15,
        // lanes 16-31 hold K=16..31 (per 32-wide K step).
        const __bf16* brow_h = &Bbuf[b][P][0][l16][half * 16];
        const __bf16* brow_l = &Bbuf[b][P][1][l16][half * 16];

        v8f acc = {};
        #pragma unroll
        for (int ks = 0; ks < 8; ++ks) {
            v16bf b_h, b_l;
            ((uint4*)&b_h)[0] = *(const uint4*)(brow_h + ks * 32);
            ((uint4*)&b_h)[1] = *(const uint4*)(brow_h + ks * 32 + 8);
            ((uint4*)&b_l)[0] = *(const uint4*)(brow_l + ks * 32);
            ((uint4*)&b_l)[1] = *(const uint4*)(brow_l + ks * 32 + 8);
            // dot ~= hi*hi + hi*lo + lo*hi (fp32-grade accuracy in bf16 WMMA)
            acc = wmma_bf16(a_hi[ks], b_h, acc);
            acc = wmma_bf16(a_hi[ks], b_l, acc);
            acc = wmma_bf16(a_lo[ks], b_h, acc);
        }

        // C/D layout: lanes 0-15 VGPR v -> row v; lanes 16-31 -> row v+8
        if (P == 0) {      // even wave: publish its K-half partial
            #pragma unroll
            for (int v = 0; v < 8; v++) ex[pair][v + 8 * half][l16] = acc[v];
        }
        __syncthreads();   // partials visible; all reads of buf cc&1 done

        if (P == 1) {      // odd wave: combine halves, distance, argmin
            const int code = cc * 16 + l16;
            const float cn = cnorm[code];
            #pragma unroll
            for (int v = 0; v < 8; v++) {
                float dot = acc[v] + ex[pair][v + 8 * half][l16];
                float d = cn - 2.0f * dot;  // ||z||^2 per-row const: dropped
                if (d < vmin[v]) { vmin[v] = d; vidx[v] = code; }
            }
        }
        if (wave == 0 && cc + 2 < 64) issue_chunk(cc + 2);  // refill buffer
    }

    // ---- Cross-lane argmin (odd waves) within each 16-lane half ----------
    if (P == 1) {
        #pragma unroll
        for (int m = 1; m < 16; m <<= 1) {
            #pragma unroll
            for (int v = 0; v < 8; v++) {
                float om = __shfl_xor(vmin[v], m, 32);
                int   oi = __shfl_xor(vidx[v], m, 32);
                if (om < vmin[v] || (om == vmin[v] && oi < vidx[v])) {
                    vmin[v] = om; vidx[v] = oi;
                }
            }
        }
        if (l16 == 0) {    // lanes 0 and 16: rows v and v+8 of this pair
            #pragma unroll
            for (int v = 0; v < 8; v++) {
                rowIdx[pair * 16 + v + 8 * half] = vidx[v];
            }
        }
    }
    __syncthreads();

    // ---- Phase 3: residual update r -= cb[idx]; accumulate sum sq --------
    float local = 0.f;
    for (int e = tid; e < ROWS_PER_BLOCK * RQ_C; e += THREADS) {
        int r = e >> 9, c = e & (RQ_C - 1);
        float  q = cb_f32[(size_t)rowIdx[r] * RQ_C + c];
        size_t g = (size_t)(row0 + r) * RQ_C + c;
        float nv = resid[g] - q;
        resid[g] = nv;
        local += nv * nv;
    }
    #pragma unroll
    for (int m = 16; m >= 1; m >>= 1) local += __shfl_xor(local, m, 32);
    if (lane == 0) wsum[wave] = local;
    __syncthreads();
    if (tid == 0) {
        float s = 0.f;
        #pragma unroll
        for (int w = 0; w < 8; w++) s += wsum[w];
        atomicAdd(sumsq, (double)s);
    }
}

// ---------------------------------------------------------------------------
// Finalize: quant_sum = x - final_residual; loss = 2.75 * sum(r_i^2)/(N*C)
// ---------------------------------------------------------------------------
__global__ __launch_bounds__(256) void rq_finalize_kernel(
    const float* __restrict__ x, const float* __restrict__ resid,
    float* __restrict__ out, const double* __restrict__ sumsq, size_t n) {
    size_t i = (size_t)blockIdx.x * 256 + threadIdx.x;
    if (i < n) out[i] = x[i] - resid[i];
    if (i == 0) out[n] = (float)(2.75 * (*sumsq) / (double)n);
}

extern "C" void kernel_launch(void* const* d_in, const int* in_sizes, int n_in,
                              void* d_out, int out_size, void* d_ws, size_t ws_size,
                              hipStream_t stream) {
    (void)in_sizes; (void)n_in; (void)out_size; (void)ws_size;
    const float* x  = (const float*)d_in[0];             // [B,W,C]
    const float* cb = (const float*)d_in[1];             // [Q,K,C]
    float* out = (float*)d_out;                          // [N*C] + 1 scalar

    // Workspace layout (cbh || cbl contiguous: TDM z-stride = LO_ELEMS)
    char* ws = (char*)d_ws;
    double* sumsq = (double*)ws;                                   // 8 B
    float*  cnorm = (float*)(ws + 256);                            // 32 KB
    __bf16* cbh   = (__bf16*)(ws + 256 + 32768);                   // 8 MB
    __bf16* cbl   = cbh + (size_t)LO_ELEMS;                        // 8 MB
    float*  resid = (float*)(ws + 256 + 32768 +
                             (size_t)2 * LO_ELEMS * 2);            // 64 MB

    const size_t NC = (size_t)RQ_N * RQ_C;

    // residual <- x
    (void)hipMemcpyAsync(resid, x, NC * sizeof(float),
                         hipMemcpyDeviceToDevice, stream);
    rq_zero_kernel<<<1, 1, 0, stream>>>(sumsq);

    // Split all stage codebooks to bf16 hi/lo; per-code norms
    const int ncb = RQ_Q * RQ_K * RQ_C;
    rq_split_kernel<<<(ncb + 255) / 256, 256, 0, stream>>>(cb, cbh, cbl, ncb);
    rq_norm_kernel<<<(RQ_Q * RQ_K * 32 + 255) / 256, 256, 0, stream>>>(
        cb, cnorm, RQ_Q * RQ_K);

    // Sequential quantizer stages (residual dependency)
    for (int q = 0; q < RQ_Q; ++q) {
        const size_t co = (size_t)q * RQ_K * RQ_C;
        rq_stage_kernel<<<RQ_N / ROWS_PER_BLOCK, THREADS, 0, stream>>>(
            cb + co, cbh + co, cnorm + (size_t)q * RQ_K,
            resid, sumsq);
    }

    rq_finalize_kernel<<<(unsigned)((NC + 255) / 256), 256, 0, stream>>>(
        x, resid, out, sumsq, NC);
}
